// VQVAE_13245679141538
// MI455X (gfx1250) — matrix operand order repaired and checked by reading, not compiled, not run on hardware
//
#include <hip/hip_runtime.h>
#include <hip/hip_bf16.h>
#include <stddef.h>

// ---------------------------------------------------------------------------
// VQ-VAE forward for MI455X (gfx1250, wave32).  All GEMMs via
// v_wmma_f32_16x16x32_bf16; activations staged f32 NHWC; BN+ReLU fused into
// bf16 im2col pack kernels that emit WMMA fragment order directly.
// ---------------------------------------------------------------------------

typedef __attribute__((ext_vector_type(16))) __bf16 v16bf;
typedef __attribute__((ext_vector_type(8)))  float  v8f;

#define BATCH 256
#define CH    256
#define KCB   1024

// ---------------------------------------------------------------------------
// Pack im2col patches (A matrix) into WMMA A-fragment order.
//   frag layout: dst[(mt*KT+kt)*512 + lane*16 + h]
//   A 16x32 bf16 layout: m = mt*16 + (lane&15)
//                        K = kt*32 + ((lane>=16)?16:0) + h
//   K decomposes as (tap = K/C, ci = K%C), tap = ky*KW + kx
//   Optional fused BN (v*sc[ci]+sh[ci]) and ReLU.
// ---------------------------------------------------------------------------
__global__ __launch_bounds__(256) void pack_A_kernel(
    const float* __restrict__ src, const float* __restrict__ sc,
    const float* __restrict__ sh, int relu,
    int Hin, int Win, int C, int Hout, int Wout,
    int KH, int KW, int stride, int pady, int padx,
    int KT, __bf16* __restrict__ dst)
{
  const int mt = blockIdx.x;
  const int kt = blockIdx.y;
  const int HWout = Hout * Wout;
  for (int e = threadIdx.x; e < 512; e += 256) {
    const int lane = e >> 4, h = e & 15;
    const int m  = mt * 16 + (lane & 15);
    const int K  = kt * 32 + ((lane >= 16) ? 16 : 0) + h;
    const int ci = K % C;
    const int tap = K / C;
    const int ky = tap / KW, kx = tap % KW;
    const int b  = m / HWout;
    const int p  = m % HWout;
    const int oy = p / Wout, ox = p % Wout;
    const int iy = oy * stride - pady + ky;
    const int ix = ox * stride - padx + kx;
    float v = 0.0f;
    if (iy >= 0 && iy < Hin && ix >= 0 && ix < Win)
      v = src[(((size_t)b * Hin + iy) * Win + ix) * C + ci];
    if (sc)   v = v * sc[ci] + sh[ci];
    if (relu) v = fmaxf(v, 0.0f);
    dst[(size_t)(mt * KT + kt) * 512 + e] = (__bf16)v;
  }
}

// ---------------------------------------------------------------------------
// Pack conv weights (OIHW, n = O, K = tap*Ci + ci) into WMMA B-fragment order.
//   B 32x16 bf16 layout: n = nt*16 + (lane&15)
//                        K = kt*32 + ((lane<16)?0:16) + h
// ---------------------------------------------------------------------------
__global__ __launch_bounds__(256) void pack_W_kernel(
    const float* __restrict__ w, int Ci, int KH, int KW, int KT,
    __bf16* __restrict__ dst)
{
  const int nt = blockIdx.x;
  const int kt = blockIdx.y;
  for (int e = threadIdx.x; e < 512; e += 256) {
    const int lane = e >> 4, h = e & 15;
    const int n = nt * 16 + (lane & 15);
    const int K = kt * 32 + ((lane < 16) ? 0 : 16) + h;
    const int ci = K % Ci;
    const int tap = K / Ci;
    const int ky = tap / KW, kx = tap % KW;
    dst[(size_t)(nt * KT + kt) * 512 + e] =
        (__bf16)w[(((size_t)n * Ci + ci) * KH + ky) * KW + kx];
  }
}

// dt1 transpose-conv weights per output-parity class (py,px):
//   K = t*256 + ci, t=(ty,tx), ky = 2*ty+py, kx = 2*tx+px.  KT = 32.
__global__ __launch_bounds__(256) void pack_W_dt1_kernel(
    const float* __restrict__ w, int py, int px, __bf16* __restrict__ dst)
{
  const int nt = blockIdx.x;
  const int kt = blockIdx.y;
  for (int e = threadIdx.x; e < 512; e += 256) {
    const int lane = e >> 4, h = e & 15;
    const int n = nt * 16 + (lane & 15);
    const int K = kt * 32 + ((lane < 16) ? 0 : 16) + h;
    const int ci = K & 255;
    const int t  = K >> 8;
    const int ky = 2 * (t >> 1) + py;
    const int kx = 2 * (t & 1) + px;
    dst[(size_t)(nt * 32 + kt) * 512 + e] =
        (__bf16)w[(((size_t)n * 256 + ci) * 4 + ky) * 4 + kx];
  }
}

// ---------------------------------------------------------------------------
// WMMA GEMM: 8 waves/block, each wave owns one 16(M) x 64(N) tile.
//   mode 0: out[r*C+n] = acc + bias[n] (+ skip[r*C+n]);  r = m
//   mode 1: parity scatter (transpose-conv): r = b*HWout + (2qy+py)*Wout+(2qx+px)
//   mode 2: VQ distance: out = bias[n] - 2*acc
// ---------------------------------------------------------------------------
__global__ __launch_bounds__(256) void wmma_gemm_kernel(
    const __bf16* __restrict__ Ap, const __bf16* __restrict__ Wp,
    const float* __restrict__ bias, const float* __restrict__ skip,
    float* __restrict__ out,
    int KT, int Cout, int mode,
    int HWcls, int Wcls, int Wout, int HWout, int py, int px)
{
  const int lane  = threadIdx.x & 31;
  const int wave  = threadIdx.x >> 5;
  const int mtile = blockIdx.x * 8 + wave;
  const int nt0   = blockIdx.y * 4;

  const v16bf* Af = (const v16bf*)Ap + (size_t)mtile * KT * 32 + lane;
  const v16bf* B0 = (const v16bf*)Wp + (size_t)(nt0 + 0) * KT * 32 + lane;
  const v16bf* B1 = (const v16bf*)Wp + (size_t)(nt0 + 1) * KT * 32 + lane;
  const v16bf* B2 = (const v16bf*)Wp + (size_t)(nt0 + 2) * KT * 32 + lane;
  const v16bf* B3 = (const v16bf*)Wp + (size_t)(nt0 + 3) * KT * 32 + lane;

  v8f z = {0.f, 0.f, 0.f, 0.f, 0.f, 0.f, 0.f, 0.f};
  v8f acc[4] = {z, z, z, z};

  for (int kt = 0; kt < KT; ++kt) {
    const v16bf a  = Af[(size_t)kt * 32];
    const v16bf b0 = B0[(size_t)kt * 32];
    const v16bf b1 = B1[(size_t)kt * 32];
    const v16bf b2 = B2[(size_t)kt * 32];
    const v16bf b3 = B3[(size_t)kt * 32];
    acc[0] = __builtin_amdgcn_wmma_f32_16x16x32_bf16(false, a, false, b0,
                                                     (short)0, acc[0], false, false);
    acc[1] = __builtin_amdgcn_wmma_f32_16x16x32_bf16(false, a, false, b1,
                                                     (short)0, acc[1], false, false);
    acc[2] = __builtin_amdgcn_wmma_f32_16x16x32_bf16(false, a, false, b2,
                                                     (short)0, acc[2], false, false);
    acc[3] = __builtin_amdgcn_wmma_f32_16x16x32_bf16(false, a, false, b3,
                                                     (short)0, acc[3], false, false);
  }

  // Epilogue per C/D layout: lane<16 -> M=v, lane>=16 -> M=v+8; N = lane&15.
  const int ncol = lane & 15;
  for (int j = 0; j < 4; ++j) {
    const int n = (nt0 + j) * 16 + ncol;
    const float bv = bias ? bias[n] : 0.0f;
    union { v8f v; float f[8]; } u;
    u.v = acc[j];
    for (int v = 0; v < 8; ++v) {
      const int mrow = (lane < 16) ? v : (v + 8);
      const int m = mtile * 16 + mrow;
      size_t r;
      if (mode == 1) {
        const int b  = m / HWcls;
        const int pp = m % HWcls;
        const int qy = pp / Wcls, qx = pp % Wcls;
        r = (size_t)b * HWout + (size_t)(2 * qy + py) * Wout + (2 * qx + px);
      } else {
        r = (size_t)m;
      }
      float val;
      if (mode == 2) {
        val = bv - 2.0f * u.f[v];
      } else {
        val = u.f[v] + bv;
        if (skip) val += skip[r * Cout + n];
      }
      out[r * (size_t)Cout + n] = val;
    }
  }
}

// ---------------------------------------------------------------------------
// BatchNorm (training mode) statistics: deterministic two-stage reduction.
// ---------------------------------------------------------------------------
__global__ __launch_bounds__(256) void bn_part1_kernel(
    const float* __restrict__ act, float* __restrict__ psum,
    float* __restrict__ psq)
{
  const int c = threadIdx.x;                 // channel
  const int chunk = blockIdx.x;              // 256 rows per chunk
  const size_t base = (size_t)chunk * 256 * 256 + c;
  float s = 0.0f, q = 0.0f;
  for (int r = 0; r < 256; ++r) {
    const float v = act[base + (size_t)r * 256];
    s += v; q += v * v;
  }
  psum[chunk * 256 + c] = s;
  psq [chunk * 256 + c] = q;
}

__global__ __launch_bounds__(256) void bn_part2_kernel(
    const float* __restrict__ psum, const float* __restrict__ psq,
    int nchunks, int NHW, const float* __restrict__ g,
    const float* __restrict__ b, float* __restrict__ sc,
    float* __restrict__ sh)
{
  const int c = threadIdx.x;
  float s = 0.0f, q = 0.0f;
  for (int k = 0; k < nchunks; ++k) {
    s += psum[k * 256 + c];
    q += psq [k * 256 + c];
  }
  const float inv = 1.0f / (float)NHW;
  const float mu  = s * inv;
  const float var = q * inv - mu * mu;
  const float sca = g[c] * rsqrtf(var + 1e-5f);
  sc[c] = sca;
  sh[c] = b[c] - mu * sca;
}

// ---------------------------------------------------------------------------
// e1: 3->256, k=4, s=2, p=1 on 32x32 input (NHWC) -> a16 (B,16,16,256)
// Direct VALU conv, one block per output pixel, thread = out channel.
// ---------------------------------------------------------------------------
__global__ __launch_bounds__(256) void conv_e1_kernel(
    const float* __restrict__ x, const float* __restrict__ w,
    const float* __restrict__ bias, float* __restrict__ out)
{
  const int o   = threadIdx.x;
  const int pix = blockIdx.x;          // b*256 + oy*16 + ox
  const int b = pix >> 8, p = pix & 255;
  const int oy = p >> 4, ox = p & 15;
  float acc = bias[o];
  for (int ky = 0; ky < 4; ++ky) {
    const int iy = 2 * oy - 1 + ky;
    if (iy < 0 || iy >= 32) continue;
    for (int kx = 0; kx < 4; ++kx) {
      const int ix = 2 * ox - 1 + kx;
      if (ix < 0 || ix >= 32) continue;
      const float* xp = x + (((size_t)b * 32 + iy) * 32 + ix) * 3;
      const int wb = o * 48 + ky * 4 + kx;   // w[((o*3+ci)*4+ky)*4+kx]
      acc += xp[0] * w[wb] + xp[1] * w[wb + 16] + xp[2] * w[wb + 32];
    }
  }
  out[(size_t)pix * 256 + o] = acc;
}

// ||c_n||^2 per codebook row
__global__ __launch_bounds__(256) void cnorm_kernel(
    const float* __restrict__ cb, float* __restrict__ cn)
{
  __shared__ float red[256];
  const int n = blockIdx.x;
  const float v = cb[(size_t)n * 256 + threadIdx.x];
  red[threadIdx.x] = v * v;
  __syncthreads();
  for (int s = 128; s > 0; s >>= 1) {
    if (threadIdx.x < s) red[threadIdx.x] += red[threadIdx.x + s];
    __syncthreads();
  }
  if (threadIdx.x == 0) cn[n] = red[0];
}

// argmin over 1024 codes per spatial position (first-occurrence tie-break)
__global__ __launch_bounds__(256) void argmin_kernel(
    const float* __restrict__ dist, int* __restrict__ qi,
    float* __restrict__ qf)
{
  __shared__ float sv[256];
  __shared__ int   si[256];
  const int m = blockIdx.x;
  const int t = threadIdx.x;
  float best = 3.4e38f; int bi = 0;
  for (int n = t; n < 1024; n += 256) {
    const float v = dist[(size_t)m * 1024 + n];
    if (v < best) { best = v; bi = n; }
  }
  sv[t] = best; si[t] = bi;
  __syncthreads();
  for (int s = 128; s > 0; s >>= 1) {
    if (t < s) {
      if (sv[t + s] < sv[t] || (sv[t + s] == sv[t] && si[t + s] < si[t])) {
        sv[t] = sv[t + s]; si[t] = si[t + s];
      }
    }
    __syncthreads();
  }
  if (t == 0) { qi[m] = si[0]; qf[m] = (float)si[0]; }
}

__global__ __launch_bounds__(256) void gather_zq_kernel(
    const int* __restrict__ qi, const float* __restrict__ cb,
    float* __restrict__ zq)
{
  const int m = blockIdx.x;
  const int c = threadIdx.x;
  zq[(size_t)m * 256 + c] = cb[(size_t)qi[m] * 256 + c];
}

// sum((a-b)^2) -> per-block partials (deterministic)
__global__ __launch_bounds__(256) void sqdiff_kernel(
    const float* __restrict__ a, const float* __restrict__ b,
    unsigned int n, float* __restrict__ partial)
{
  __shared__ float red[256];
  float s = 0.0f;
  for (unsigned int i = blockIdx.x * 256 + threadIdx.x; i < n;
       i += gridDim.x * 256) {
    const float d = a[i] - b[i];
    s += d * d;
  }
  red[threadIdx.x] = s;
  __syncthreads();
  for (int k = 128; k > 0; k >>= 1) {
    if (threadIdx.x < k) red[threadIdx.x] += red[threadIdx.x + k];
    __syncthreads();
  }
  if (threadIdx.x == 0) partial[blockIdx.x] = red[0];
}

// recon partials: outs is NCHW (B,3,32,32), x is NHWC (B,32,32,3)
__global__ __launch_bounds__(256) void sqdiff_recon_kernel(
    const float* __restrict__ outs, const float* __restrict__ x,
    float* __restrict__ partial)
{
  __shared__ float red[256];
  float s = 0.0f;
  for (unsigned int g = blockIdx.x * 256 + threadIdx.x; g < 786432u;
       g += gridDim.x * 256) {
    const int ox = g & 31;
    const int oy = (g >> 5) & 31;
    const int o  = (g >> 10) % 3;
    const int b  = g / 3072;
    const float xv = x[(((size_t)b * 32 + oy) * 32 + ox) * 3 + o];
    const float d = outs[g] - xv;
    s += d * d;
  }
  red[threadIdx.x] = s;
  __syncthreads();
  for (int k = 128; k > 0; k >>= 1) {
    if (threadIdx.x < k) red[threadIdx.x] += red[threadIdx.x + k];
    __syncthreads();
  }
  if (threadIdx.x == 0) partial[blockIdx.x] = red[0];
}

__global__ __launch_bounds__(256) void final_reduce_kernel(
    const float* __restrict__ partial, int n, float scale,
    float* __restrict__ d0, float* __restrict__ d1)
{
  __shared__ float red[256];
  float s = 0.0f;
  for (int i = threadIdx.x; i < n; i += 256) s += partial[i];
  red[threadIdx.x] = s;
  __syncthreads();
  for (int k = 128; k > 0; k >>= 1) {
    if (threadIdx.x < k) red[threadIdx.x] += red[threadIdx.x + k];
    __syncthreads();
  }
  if (threadIdx.x == 0) {
    const float r = red[0] * scale;
    *d0 = r;
    if (d1) *d1 = r;
  }
}

// ---------------------------------------------------------------------------
// dt2: transpose conv 256->3 (lhs-dilated conv, k=4, pad=2) with fused
// d_bn2 + ReLU.  One block per output pixel (b,oy,ox); thread = input channel;
// 3 output channels reduced through LDS.  Writes outs NCHW.
// ---------------------------------------------------------------------------
__global__ __launch_bounds__(256) void dt2_kernel(
    const float* __restrict__ h16, const float* __restrict__ sc,
    const float* __restrict__ sh, const float* __restrict__ w,
    const float* __restrict__ bias, float* __restrict__ outs)
{
  __shared__ float red[256];
  const int blk = blockIdx.x;          // b*1024 + oy*32 + ox
  const int b  = blk >> 10;
  const int p  = blk & 1023;
  const int oy = p >> 5, ox = p & 31;
  const int t  = threadIdx.x;          // ci
  const int pyr = oy & 1, pxr = ox & 1;
  const int qy = oy >> 1, qx = ox >> 1;

  float a0 = 0.0f, a1 = 0.0f, a2 = 0.0f;
  for (int ty = 0; ty < 2; ++ty) {
    const int iy = qy + ty + pyr - 1;
    if (iy < 0 || iy >= 16) continue;
    const int ky = 2 * ty + pyr;
    for (int tx = 0; tx < 2; ++tx) {
      const int ix = qx + tx + pxr - 1;
      if (ix < 0 || ix >= 16) continue;
      const int kx = 2 * tx + pxr;
      const float hv =
          fmaxf(sc[t] * h16[(((size_t)b * 16 + iy) * 16 + ix) * 256 + t] + sh[t],
                0.0f);
      const int wb = (t * 4 + ky) * 4 + kx;   // w[((o*256+ci)*4+ky)*4+kx]
      a0 += hv * w[wb];
      a1 += hv * w[wb + 4096];
      a2 += hv * w[wb + 8192];
    }
  }
  float accs[3] = {a0, a1, a2};
  for (int o = 0; o < 3; ++o) {
    red[t] = accs[o];
    __syncthreads();
    for (int k = 128; k > 0; k >>= 1) {
      if (t < k) red[t] += red[t + k];
      __syncthreads();
    }
    if (t == 0)
      outs[(((size_t)b * 3 + o) * 32 + oy) * 32 + ox] = red[0] + bias[o];
    __syncthreads();
  }
}

// ---------------------------------------------------------------------------
// Host orchestration
// ---------------------------------------------------------------------------
namespace {

struct Ws {
  float*  a16;     // 67.1 MB: (B,16,16,256) f32; reused as dist (16384x1024)
  float*  a8_0;    // 16.8 MB
  float*  a8_1;    // 16.8 MB
  float*  a8_2;    // 16.8 MB
  __bf16* Ap;      // 134.2 MB packed A fragments
  __bf16* Wp;      // 2 MB packed W fragments
  float*  sc;      // 256
  float*  sh;      // 256
  float*  cnorm;   // 1024
  int*    qidx;    // 16384
  float*  bn_sum;  // 65536
  float*  bn_sq;   // 65536
  float*  lpart;   // 1024
};

inline Ws carve(void* ws) {
  char* c = (char*)ws;
  Ws w;
  size_t off = 0;
  w.a16   = (float*) (c + off); off += 67108864;
  w.a8_0  = (float*) (c + off); off += 16777216;
  w.a8_1  = (float*) (c + off); off += 16777216;
  w.a8_2  = (float*) (c + off); off += 16777216;
  w.Ap    = (__bf16*)(c + off); off += 134217728;
  w.Wp    = (__bf16*)(c + off); off += 2097152;
  w.sc    = (float*) (c + off); off += 1024;
  w.sh    = (float*) (c + off); off += 1024;
  w.cnorm = (float*) (c + off); off += 4096;
  w.qidx  = (int*)   (c + off); off += 65536;
  w.bn_sum= (float*) (c + off); off += 262144;
  w.bn_sq = (float*) (c + off); off += 262144;
  w.lpart = (float*) (c + off); off += 4096;
  return w;
}

inline void run_bn(const float* act, int NHW, const float* g, const float* b,
                   const Ws& w, hipStream_t s)
{
  const int chunks = NHW / 256;
  bn_part1_kernel<<<chunks, 256, 0, s>>>(act, w.bn_sum, w.bn_sq);
  bn_part2_kernel<<<1, 256, 0, s>>>(w.bn_sum, w.bn_sq, chunks, NHW, g, b,
                                    w.sc, w.sh);
}

// Conv with fused BN+ReLU input, C=256 in/out. (3x3 res conv or 4x4 e2 conv)
inline void run_conv(const float* src, const float* wgt, const float* bias,
                     const float* skip, float* out, int Hin, int Hout, int KH,
                     int stride, int pad, const Ws& w, hipStream_t s)
{
  const int KT = KH * KH * 256 / 32;
  const int Mtiles = BATCH * Hout * Hout / 16;
  pack_A_kernel<<<dim3(Mtiles, KT), 256, 0, s>>>(
      src, w.sc, w.sh, 1, Hin, Hin, 256, Hout, Hout, KH, KH, stride, pad, pad,
      KT, w.Ap);
  pack_W_kernel<<<dim3(16, KT), 256, 0, s>>>(wgt, 256, KH, KH, KT, w.Wp);
  wmma_gemm_kernel<<<dim3(Mtiles / 8, 4), 256, 0, s>>>(
      w.Ap, w.Wp, bias, skip, out, KT, 256, 0, 0, 0, 0, 0, 0, 0);
}

// Residual block at 8x8: X -> Y using T as temp.  base = index of "<n>_bn1_g".
inline void run_res(void* const* din, int base, const float* X, float* T,
                    float* Y, const Ws& w, hipStream_t s)
{
  const float* bn1_g = (const float*)din[base + 0];
  const float* bn1_b = (const float*)din[base + 1];
  const float* c1_w  = (const float*)din[base + 2];
  const float* c1_b  = (const float*)din[base + 3];
  const float* bn2_g = (const float*)din[base + 4];
  const float* bn2_b = (const float*)din[base + 5];
  const float* c2_w  = (const float*)din[base + 6];
  const float* c2_b  = (const float*)din[base + 7];

  run_bn(X, 16384, bn1_g, bn1_b, w, s);
  run_conv(X, c1_w, c1_b, nullptr, T, 8, 8, 3, 1, 1, w, s);
  run_bn(T, 16384, bn2_g, bn2_b, w, s);
  run_conv(T, c2_w, c2_b, X /*skip*/, Y, 8, 8, 3, 1, 1, w, s);
}

} // namespace

extern "C" void kernel_launch(void* const* d_in, const int* in_sizes, int n_in,
                              void* d_out, int out_size, void* d_ws,
                              size_t ws_size, hipStream_t stream)
{
  (void)in_sizes; (void)n_in; (void)out_size; (void)ws_size;
  Ws w = carve(d_ws);

  const float* x      = (const float*)d_in[0];
  const float* e1_w   = (const float*)d_in[1];
  const float* e1_b   = (const float*)d_in[2];
  const float* ebn1_g = (const float*)d_in[3];
  const float* ebn1_b = (const float*)d_in[4];
  const float* e2_w   = (const float*)d_in[5];
  const float* e2_b   = (const float*)d_in[6];
  // er1: 7..14, er2: 15..22, dr1: 23..30, dr2: 31..38
  const float* dbn1_g = (const float*)d_in[39];
  const float* dbn1_b = (const float*)d_in[40];
  const float* dt1_w  = (const float*)d_in[41];
  const float* dt1_b  = (const float*)d_in[42];
  const float* dbn2_g = (const float*)d_in[43];
  const float* dbn2_b = (const float*)d_in[44];
  const float* dt2_w  = (const float*)d_in[45];
  const float* dt2_b  = (const float*)d_in[46];
  const float* cb     = (const float*)d_in[47];

  float* out_scalars = (float*)d_out;              // [0]=recon [1]=vq [2]=commit
  float* out_qidx    = out_scalars + 3;            // 16384 (indices as float)
  float* out_imgs    = out_scalars + 3 + 16384;    // 786432, NCHW

  // ---- Encoder ----
  // e1 conv: x (NHWC 32x32x3) -> a16 (B,16,16,256)
  conv_e1_kernel<<<BATCH * 256, 256, 0, stream>>>(x, e1_w, e1_b, w.a16);
  // relu(bn(a16)) fused into e2 pack; e2 conv 4x4 s2 p1 -> a8_0 (ze pre-res)
  run_bn(w.a16, 65536, ebn1_g, ebn1_b, w, stream);
  run_conv(w.a16, e2_w, e2_b, nullptr, w.a8_0, 16, 8, 4, 2, 1, w, stream);
  // residual blocks
  run_res(d_in, 7,  w.a8_0, w.a8_1, w.a8_2, w, stream);   // er1
  run_res(d_in, 15, w.a8_2, w.a8_1, w.a8_0, w, stream);   // er2 -> ze in a8_0

  // ---- Vector quantization ----
  // A = ze rows (K=256), no BN/ReLU
  pack_A_kernel<<<dim3(1024, 8), 256, 0, stream>>>(
      w.a8_0, nullptr, nullptr, 0, 8, 8, 256, 8, 8, 1, 1, 1, 0, 0, 8, w.Ap);
  // B = codebook (N=1024 codes, K=256)
  pack_W_kernel<<<dim3(64, 8), 256, 0, stream>>>(cb, 256, 1, 1, 8, w.Wp);
  cnorm_kernel<<<1024, 256, 0, stream>>>(cb, w.cnorm);
  // dist[m,n] = ||c_n||^2 - 2 z.c  (||z||^2 irrelevant for argmin) -> a16
  wmma_gemm_kernel<<<dim3(128, 16), 256, 0, stream>>>(
      w.Ap, w.Wp, w.cnorm, nullptr, w.a16, 8, 1024, 2, 0, 0, 0, 0, 0, 0);
  argmin_kernel<<<16384, 256, 0, stream>>>(w.a16, w.qidx, out_qidx);
  gather_zq_kernel<<<16384, 256, 0, stream>>>(w.qidx, cb, w.a8_2); // zq
  // vq == commit (forward values): mean((ze - zq)^2)
  sqdiff_kernel<<<1024, 256, 0, stream>>>(w.a8_0, w.a8_2, 4194304u, w.lpart);
  final_reduce_kernel<<<1, 256, 0, stream>>>(w.lpart, 1024, 1.0f / 4194304.0f,
                                             out_scalars + 1, out_scalars + 2);

  // ---- Decoder ----
  run_res(d_in, 23, w.a8_2, w.a8_1, w.a8_0, w, stream);   // dr1 (input zq)
  run_res(d_in, 31, w.a8_0, w.a8_1, w.a8_2, w, stream);   // dr2 -> a8_2

  // dt1 transpose conv: relu(bn(a8_2)) -> a16 (B,16,16,256), 4 parity classes
  run_bn(w.a8_2, 16384, dbn1_g, dbn1_b, w, stream);
  for (int py = 0; py < 2; ++py) {
    for (int px = 0; px < 2; ++px) {
      pack_A_kernel<<<dim3(1024, 32), 256, 0, stream>>>(
          w.a8_2, w.sc, w.sh, 1, 8, 8, 256, 8, 8, 2, 2, 1, 1 - py, 1 - px,
          32, w.Ap);
      pack_W_dt1_kernel<<<dim3(16, 32), 256, 0, stream>>>(dt1_w, py, px, w.Wp);
      wmma_gemm_kernel<<<dim3(128, 4), 256, 0, stream>>>(
          w.Ap, w.Wp, dt1_b, nullptr, w.a16, 32, 256, 1,
          64 /*HWcls*/, 8 /*Wcls*/, 16 /*Wout*/, 256 /*HWout*/, py, px);
    }
  }

  // dt2: relu(bn(a16)) -> outs (B,3,32,32), fused BN in kernel
  run_bn(w.a16, 65536, dbn2_g, dbn2_b, w, stream);
  dt2_kernel<<<BATCH * 1024, 256, 0, stream>>>(w.a16, w.sc, w.sh, dt2_w, dt2_b,
                                               out_imgs);

  // recon = mean((x - outs)^2)
  sqdiff_recon_kernel<<<1024, 256, 0, stream>>>(out_imgs, x, w.lpart);
  final_reduce_kernel<<<1, 256, 0, stream>>>(w.lpart, 1024, 1.0f / 786432.0f,
                                             out_scalars + 0, nullptr);
}
